// SigmoidalMultiHeadAttention_10883447128523
// MI455X (gfx1250) — matrix operand order repaired and checked by reading, not compiled, not run on hardware
//
#include <hip/hip_runtime.h>
#include <hip/hip_bf16.h>
#include <math.h>

// ---------------------------------------------------------------------------
// Types for CDNA5 WMMA (wave32): D(16x16 f32) = A(16x32 f16) x B(32x16 f16) + C
// ---------------------------------------------------------------------------
typedef __attribute__((ext_vector_type(16))) _Float16 v16h;
typedef __attribute__((ext_vector_type(8)))  _Float16 v8h;
typedef __attribute__((ext_vector_type(4)))  _Float16 v4h;
typedef __attribute__((ext_vector_type(8)))  float    v8f;

#define BB 8
#define LL 1024
#define DD 1024
#define HH 16
#define DKV 64

__device__ __forceinline__ v8f wmma_f16(v16h a, v16h b, v8f c) {
    return __builtin_amdgcn_wmma_f32_16x16x32_f16(
        /*neg_a=*/false, a, /*neg_b=*/false, b,
        /*c_mod=*/(short)0, c, /*reuse_a=*/false, /*reuse_b=*/false);
}

// A-matrix fragment (16x32, row-major source, leading dim ld in halves).
// Lane r = lane&15, half = lane>>4. elems 0..7 -> k = 8*half+i ; 8..15 -> k = 16+8*half+(i-8).
__device__ __forceinline__ v16h load_fragA(const _Float16* __restrict__ base, int ld) {
    const int lane = threadIdx.x & 31;
    const int r    = lane & 15;
    const int half = lane >> 4;
    const _Float16* p = base + (size_t)r * ld + half * 8;
    v8h lo = *reinterpret_cast<const v8h*>(p);
    v8h hi = *reinterpret_cast<const v8h*>(p + 16);
    v16h f;
#pragma unroll
    for (int i = 0; i < 8; ++i) { f[i] = lo[i]; f[i + 8] = hi[i]; }
    return f;
}

// B-matrix fragment (32x16). Source stored column-contiguous: base points to a
// [16 columns][ld] array where row n holds column n of B (k contiguous).
// Lane n = lane&15, group g = lane>>4; elems 0..7 -> k = 8*g+i ; 8..15 -> k = 16+8*g+(i-8).
__device__ __forceinline__ v16h load_fragB(const _Float16* __restrict__ base, int ld) {
    const int lane = threadIdx.x & 31;
    const int n = lane & 15;
    const int g = lane >> 4;
    const _Float16* p = base + (size_t)n * ld + 8 * g;
    v8h lo = *reinterpret_cast<const v8h*>(p);
    v8h hi = *reinterpret_cast<const v8h*>(p + 16);
    v16h f;
#pragma unroll
    for (int i = 0; i < 8; ++i) { f[i] = lo[i]; f[i + 8] = hi[i]; }
    return f;
}

// ---------------------------------------------------------------------------
// fp32 -> fp16 conversion, 4 elements/thread, vectorized
// ---------------------------------------------------------------------------
__global__ void f32_to_f16_kernel(const float* __restrict__ src,
                                  _Float16* __restrict__ dst, int n) {
    int i = (blockIdx.x * blockDim.x + threadIdx.x) * 4;
    if (i >= n) return;
    float4 f = *reinterpret_cast<const float4*>(src + i);
    v4h h;
    h[0] = (_Float16)f.x; h[1] = (_Float16)f.y;
    h[2] = (_Float16)f.z; h[3] = (_Float16)f.w;
    *reinterpret_cast<v4h*>(dst + i) = h;
}

// ---------------------------------------------------------------------------
// WMMA GEMM: C[M,N] = A[M,K] @ B^T, where B is stored [N,K] (nn.Linear weight).
// Block tile 128x128, 8 waves (4x2), wave tile 32x64 (2x4 WMMA subtiles).
// Output modes: f16 (scaled), f16 transposed ([N,M], packed b128 stores),
// or f32 with residual add.
// ---------------------------------------------------------------------------
__global__ void gemm128_wmma(const _Float16* __restrict__ A,
                             const _Float16* __restrict__ B,
                             int M, int N, int K, float scale,
                             _Float16* __restrict__ Cf16, int transposed,
                             float* __restrict__ Cf32,
                             const float* __restrict__ residual) {
    const int lane = threadIdx.x & 31;
    const int r    = lane & 15;
    const int half = lane >> 4;
    const int wave = threadIdx.x >> 5;
    const int bm = blockIdx.x * 128 + (wave >> 1) * 32;   // wave M origin
    const int bn = blockIdx.y * 128 + (wave & 1) * 64;    // wave N origin

    v8f acc[2][4];
#pragma unroll
    for (int i = 0; i < 2; ++i)
#pragma unroll
        for (int j = 0; j < 4; ++j) acc[i][j] = (v8f){0.f,0.f,0.f,0.f,0.f,0.f,0.f,0.f};

    const _Float16* Abase = A + (size_t)bm * K;
    const _Float16* Bbase = B + (size_t)bn * K;

    for (int k0 = 0; k0 < K; k0 += 32) {
        v16h a0 = load_fragA(Abase + k0, K);
        v16h a1 = load_fragA(Abase + (size_t)16 * K + k0, K);
        v16h b0 = load_fragB(Bbase + (size_t)0  * K + k0, K);
        v16h b1 = load_fragB(Bbase + (size_t)16 * K + k0, K);
        v16h b2 = load_fragB(Bbase + (size_t)32 * K + k0, K);
        v16h b3 = load_fragB(Bbase + (size_t)48 * K + k0, K);
        acc[0][0] = wmma_f16(a0, b0, acc[0][0]);
        acc[0][1] = wmma_f16(a0, b1, acc[0][1]);
        acc[0][2] = wmma_f16(a0, b2, acc[0][2]);
        acc[0][3] = wmma_f16(a0, b3, acc[0][3]);
        acc[1][0] = wmma_f16(a1, b0, acc[1][0]);
        acc[1][1] = wmma_f16(a1, b1, acc[1][1]);
        acc[1][2] = wmma_f16(a1, b2, acc[1][2]);
        acc[1][3] = wmma_f16(a1, b3, acc[1][3]);
    }

    if (Cf32) {
#pragma unroll
        for (int i = 0; i < 2; ++i)
#pragma unroll
            for (int j = 0; j < 4; ++j)
#pragma unroll
                for (int v = 0; v < 8; ++v) {
                    int m = bm + i * 16 + 8 * half + v;
                    int n = bn + j * 16 + r;
                    size_t idx = (size_t)m * N + n;
                    Cf32[idx] = acc[i][j][v] + residual[idx];
                }
    } else if (transposed) {
        // store C^T into [N, M]; per lane the 8 accum rows are contiguous m
#pragma unroll
        for (int i = 0; i < 2; ++i)
#pragma unroll
            for (int j = 0; j < 4; ++j) {
                int n  = bn + j * 16 + r;
                int mb = bm + i * 16 + 8 * half;
                v8h pk;
#pragma unroll
                for (int v = 0; v < 8; ++v) pk[v] = (_Float16)(acc[i][j][v] * scale);
                *reinterpret_cast<v8h*>(Cf16 + (size_t)n * M + mb) = pk;
            }
    } else {
#pragma unroll
        for (int i = 0; i < 2; ++i)
#pragma unroll
            for (int j = 0; j < 4; ++j)
#pragma unroll
                for (int v = 0; v < 8; ++v) {
                    int m = bm + i * 16 + 8 * half + v;
                    int n = bn + j * 16 + r;
                    Cf16[(size_t)m * N + n] = (_Float16)(acc[i][j][v] * scale);
                }
    }
}

// ---------------------------------------------------------------------------
// Sigmoid attention: per block = (b, h, 64 query rows); 4 waves x 16 q-rows.
// Q pre-scaled by 1/sqrt(dk). Streams Lk in chunks of 32:
//   S = Q_tile @ K_chunk^T  (2 tiles x 2 k-steps of WMMA)
//   P = mask ? 0 : sigmoid(S)      -> written f32 to attn output
//   P (C-layout) -> LDS -> A-layout fragment
//   O += P @ V_chunk         (4 WMMAs against transposed-V columns)
// ---------------------------------------------------------------------------
__global__ void attn_sigmoid_kernel(const _Float16* __restrict__ Qh,
                                    const _Float16* __restrict__ Kh,
                                    const _Float16* __restrict__ Vt,
                                    const int* __restrict__ mask,
                                    float* __restrict__ attn,
                                    _Float16* __restrict__ ctx16) {
    __shared__ __align__(16) _Float16 lds[4][16][32];
    const int lane = threadIdx.x & 31;
    const int r    = lane & 15;
    const int half = lane >> 4;
    const int wave = threadIdx.x >> 5;

    const int blk = blockIdx.x;
    const int lqb = blk & 15;          // 16 blocks of 64 q-rows
    const int bh  = blk >> 4;
    const int h   = bh & (HH - 1);
    const int b   = bh >> 4;
    const int lq  = lqb * 64 + wave * 16;

    const _Float16* Qbase = Qh + ((size_t)(b * LL + lq) * DD) + h * DKV;
    v16h aq0 = load_fragA(Qbase,      DD);   // d = 0..31
    v16h aq1 = load_fragA(Qbase + 32, DD);   // d = 32..63

    v8f acc[4];
#pragma unroll
    for (int t = 0; t < 4; ++t) acc[t] = (v8f){0.f,0.f,0.f,0.f,0.f,0.f,0.f,0.f};

    float* attn_row = attn + ((size_t)(b * HH + h) * LL + lq) * LL;
    const int* mrow = mask + (size_t)b * LL;

    for (int kk = 0; kk < LL; kk += 32) {
        v8f s[2];
#pragma unroll
        for (int t = 0; t < 2; ++t) {
            s[t] = (v8f){0.f,0.f,0.f,0.f,0.f,0.f,0.f,0.f};
            const _Float16* Kbase = Kh + ((size_t)(b * LL + kk + t * 16) * DD) + h * DKV;
            v16h bk0 = load_fragB(Kbase,      DD);
            v16h bk1 = load_fragB(Kbase + 32, DD);
            s[t] = wmma_f16(aq0, bk0, s[t]);
            s[t] = wmma_f16(aq1, bk1, s[t]);
        }

        __syncthreads();   // prior iteration's LDS reads complete
#pragma unroll
        for (int t = 0; t < 2; ++t) {
            int col = kk + t * 16 + r;                 // key position (per-lane)
            bool masked = (mrow[col] == -999);
#pragma unroll
            for (int v = 0; v < 8; ++v) {
                int m = 8 * half + v;
                float p = masked ? 0.0f : 1.0f / (1.0f + __expf(-s[t][v]));
                attn_row[(size_t)m * LL + col] = p;
                lds[wave][m][t * 16 + r] = (_Float16)p;
            }
        }
        __syncthreads();   // LDS writes visible before A-fragment reads

        v16h pf = load_fragA(&lds[wave][0][0], 32);
#pragma unroll
        for (int t2 = 0; t2 < 4; ++t2) {
            const _Float16* Vbase = Vt + ((size_t)(h * DKV + t2 * 16) * (BB * LL)) + b * LL + kk;
            v16h bv = load_fragB(Vbase, BB * LL);
            acc[t2] = wmma_f16(pf, bv, acc[t2]);
        }
    }

    // store context [B*L, H*DV] f16
#pragma unroll
    for (int t2 = 0; t2 < 4; ++t2)
#pragma unroll
        for (int v = 0; v < 8; ++v) {
            int m = lq + 8 * half + v;
            int n = h * DKV + t2 * 16 + r;
            ctx16[(size_t)(b * LL + m) * DD + n] = (_Float16)acc[t2][v];
        }
}

// ---------------------------------------------------------------------------
// LayerNorm over D=1024 per row; 256 threads, 4 elems/thread
// ---------------------------------------------------------------------------
__global__ void layernorm_kernel(const float* __restrict__ x,
                                 const float* __restrict__ gamma,
                                 const float* __restrict__ beta,
                                 float* __restrict__ out) {
    __shared__ float s1[256];
    __shared__ float s2[256];
    const int row = blockIdx.x;
    const int tid = threadIdx.x;
    const float* xr = x + (size_t)row * DD;
    float v[4], a = 0.f, q = 0.f;
#pragma unroll
    for (int i = 0; i < 4; ++i) {
        v[i] = xr[tid + i * 256];
        a += v[i];
        q += v[i] * v[i];
    }
    s1[tid] = a; s2[tid] = q;
    __syncthreads();
    for (int off = 128; off > 0; off >>= 1) {
        if (tid < off) { s1[tid] += s1[tid + off]; s2[tid] += s2[tid + off]; }
        __syncthreads();
    }
    float mean = s1[0] * (1.0f / DD);
    float var  = s2[0] * (1.0f / DD) - mean * mean;
    float rstd = rsqrtf(var + 1e-6f);
#pragma unroll
    for (int i = 0; i < 4; ++i) {
        int c = tid + i * 256;
        out[(size_t)row * DD + c] = (v[i] - mean) * rstd * gamma[c] + beta[c];
    }
}

// ---------------------------------------------------------------------------
// Host launch
// ---------------------------------------------------------------------------
extern "C" void kernel_launch(void* const* d_in, const int* in_sizes, int n_in,
                              void* d_out, int out_size, void* d_ws, size_t ws_size,
                              hipStream_t stream) {
    const float* query = (const float*)d_in[0];
    const float* key_i = (const float*)d_in[1];
    const float* val_i = (const float*)d_in[2];
    const int*   mask  = (const int*)d_in[3];
    const float* Wq    = (const float*)d_in[4];
    const float* Wk    = (const float*)d_in[5];
    const float* Wv    = (const float*)d_in[6];
    const float* Wo    = (const float*)d_in[7];
    const float* gamma = (const float*)d_in[8];
    const float* beta  = (const float*)d_in[9];

    const size_t NX = (size_t)BB * LL * DD;   // 8388608
    const size_t NW = (size_t)DD * DD;        // 1048576

    float* out  = (float*)d_out;              // [B,L,D]
    float* attn = out + NX;                   // [B,H,L,L]

    _Float16* w = (_Float16*)d_ws;
    _Float16* Xq16 = w;                 // inputs in f16
    _Float16* Xk16 = Xq16 + NX;
    _Float16* Xv16 = Xk16 + NX;
    _Float16* Wq16 = Xv16 + NX;
    _Float16* Wk16 = Wq16 + NW;
    _Float16* Wv16 = Wk16 + NW;
    _Float16* Wo16 = Wv16 + NW;
    _Float16* Qh   = Wo16 + NW;         // [B*L, H*DK], pre-scaled by 1/8
    _Float16* Kh   = Qh + NX;           // [B*L, H*DK]
    _Float16* Vt   = Kh + NX;           // transposed: [H*DV, B*L]
    _Float16* ctx  = Vt + NX;           // [B*L, H*DV]
    float*    pre  = (float*)(ctx + NX); // [B*L, D] f32 pre-LN

    // 1) convert inputs + weights to f16
    f32_to_f16_kernel<<<NX / 1024, 256, 0, stream>>>(query, Xq16, (int)NX);
    f32_to_f16_kernel<<<NX / 1024, 256, 0, stream>>>(key_i, Xk16, (int)NX);
    f32_to_f16_kernel<<<NX / 1024, 256, 0, stream>>>(val_i, Xv16, (int)NX);
    f32_to_f16_kernel<<<NW / 1024, 256, 0, stream>>>(Wq, Wq16, (int)NW);
    f32_to_f16_kernel<<<NW / 1024, 256, 0, stream>>>(Wk, Wk16, (int)NW);
    f32_to_f16_kernel<<<NW / 1024, 256, 0, stream>>>(Wv, Wv16, (int)NW);
    f32_to_f16_kernel<<<NW / 1024, 256, 0, stream>>>(Wo, Wo16, (int)NW);

    // 2) projections (M=8192, N=1024, K=1024)
    dim3 gg(BB * LL / 128, DD / 128);   // (64, 8)
    gemm128_wmma<<<gg, 256, 0, stream>>>(Xq16, Wq16, BB * LL, DD, DD,
                                         0.125f, Qh, 0, nullptr, nullptr);
    gemm128_wmma<<<gg, 256, 0, stream>>>(Xk16, Wk16, BB * LL, DD, DD,
                                         1.0f, Kh, 0, nullptr, nullptr);
    gemm128_wmma<<<gg, 256, 0, stream>>>(Xv16, Wv16, BB * LL, DD, DD,
                                         1.0f, Vt, 1, nullptr, nullptr);

    // 3) sigmoid attention; B*H*(L/64) = 2048 blocks, 4 waves each
    attn_sigmoid_kernel<<<BB * HH * (LL / 64), 128, 0, stream>>>(
        Qh, Kh, Vt, mask, attn, ctx);

    // 4) output projection + residual (f32 out)
    gemm128_wmma<<<gg, 256, 0, stream>>>(ctx, Wo16, BB * LL, DD, DD,
                                         1.0f, nullptr, 0, pre, query);

    // 5) layernorm
    layernorm_kernel<<<BB * LL, 256, 0, stream>>>(pre, gamma, beta, out);
}
